// ConvSSMPointTracker_19628000543430
// MI455X (gfx1250) — compile-verified, matrix operand order checked by reading, not run on hardware
//
#include <hip/hip_runtime.h>
#include <hip/hip_bf16.h>

// ---------------------------------------------------------------------------
// ConvSSMPointTracker for MI455X (gfx1250, wave32, WMMA + async/TDM staging).
//
//   conv1 (7x7x3 s2)  : fp32 VALU + LDS slab  -> X1 bf16 (16,128,128,64)
//   conv2 (3x3  s2)   : WMMA bf16 implicit GEMM -> X2 bf16 (16,64,64,128)
//   conv3 (3x3  s2)   : WMMA bf16 implicit GEMM -> X3 bf16 (16,32,32,256)
//   qfeat             : bilinear sample          (4,256)
//   corr+softmax+proj : -> CS f32 (4,16,32,32,64)
//   3x refinement     : SSM assoc-scan of constant (A,Bx) == geometric series
//                       == 8 depthwise circular 3x7x7 convs (LDS/TDM-staged)
//                       + layernorm + residual
//   heads             : heat softmax -> pred coords; pooled -> occlusion
// ---------------------------------------------------------------------------

typedef __bf16 bf16;
typedef float v8f  __attribute__((ext_vector_type(8)));
typedef bf16  v16bf __attribute__((ext_vector_type(16)));
typedef unsigned int v8u __attribute__((ext_vector_type(8)));
typedef unsigned int v4u __attribute__((ext_vector_type(4)));
typedef int v4i __attribute__((ext_vector_type(4)));
typedef int v8i __attribute__((ext_vector_type(8)));

// --- CDNA5 data-mover availability (guarded: never regress the compile) ----
#if defined(__HIP_DEVICE_COMPILE__) && __has_builtin(__builtin_amdgcn_global_load_async_to_lds_b128) && __has_builtin(__builtin_amdgcn_s_wait_asynccnt)
#define USE_ASYNC_LDS 1
#else
#define USE_ASYNC_LDS 0
#endif

#if defined(__HIP_DEVICE_COMPILE__) && __has_builtin(__builtin_amdgcn_tensor_load_to_lds) && __has_builtin(__builtin_amdgcn_s_wait_tensorcnt)
#define USE_TDM 1
#if __has_include(<hip/amd_detail/amd_gfx1250_TDM.h>)
#define TDM_SIX_ARG 1      // amdgpu-toolchain / therock headers -> 6-arg builtin
#else
#define TDM_SIX_ARG 0      // ROCm 7.2 clang-22 -> 5-arg builtin
#endif
#else
#define USE_TDM 0
#endif

#define DEVFN __device__ __forceinline__

DEVFN float block_reduce_sum(float v, volatile float* red, int tid) {
    red[tid] = v; __syncthreads();
    for (int s = 128; s > 0; s >>= 1) {
        if (tid < s) red[tid] += red[tid + s];
        __syncthreads();
    }
    float r = red[0]; __syncthreads();
    return r;
}

DEVFN float block_reduce_max(float v, volatile float* red, int tid) {
    red[tid] = v; __syncthreads();
    for (int s = 128; s > 0; s >>= 1) {
        if (tid < s) red[tid] = fmaxf(red[tid], red[tid + s]);
        __syncthreads();
    }
    float r = red[0]; __syncthreads();
    return r;
}

// ---------------------------------------------------------------------------
// conv1: (16,256,256,3)/255 -> (16,128,128,64), k=7, s=2, SAME (pad_before=2)
// ---------------------------------------------------------------------------
__global__ __launch_bounds__(256)
void conv1_kernel(const float* __restrict__ video, const float* __restrict__ w,
                  const float* __restrict__ bias, bf16* __restrict__ out) {
    __shared__ float slab[7 * 37 * 3];
    const int x0 = blockIdx.x << 4;
    const int y  = blockIdx.y;
    const int t  = blockIdx.z;
    const int tid = threadIdx.x;

    for (int i = tid; i < 7 * 37 * 3; i += 256) {
        int c   = i % 3;
        int col = (i / 3) % 37;
        int row = i / (3 * 37);
        int iy = 2 * y  + row - 2;
        int ix = 2 * x0 + col - 2;
        float v = 0.0f;
        if (iy >= 0 && iy < 256 && ix >= 0 && ix < 256)
            v = video[(((size_t)t * 256 + iy) * 256 + ix) * 3 + c] * (1.0f / 255.0f);
        slab[i] = v;
    }
    __syncthreads();

    const int m  = tid & 15;
    const int cg = tid >> 4;
    const int c0 = cg * 4;
    float acc[4] = {bias[c0], bias[c0 + 1], bias[c0 + 2], bias[c0 + 3]};

    for (int ky = 0; ky < 7; ++ky)
        for (int kx = 0; kx < 7; ++kx) {
            const float* sp = &slab[((ky * 37) + (2 * m + kx)) * 3];
            const float* wp = &w[((ky * 7 + kx) * 3) * 64 + c0];
#pragma unroll
            for (int ci = 0; ci < 3; ++ci) {
                float v = sp[ci];
                acc[0] += v * wp[ci * 64 + 0];
                acc[1] += v * wp[ci * 64 + 1];
                acc[2] += v * wp[ci * 64 + 2];
                acc[3] += v * wp[ci * 64 + 3];
            }
        }

    bf16* op = &out[(((size_t)t * 128 + y) * 128 + (x0 + m)) * 64 + c0];
#pragma unroll
    for (int j = 0; j < 4; ++j) {
        float v = acc[j] > 0.0f ? acc[j] : 0.0f;
        op[j] = (bf16)v;
    }
}

// ---------------------------------------------------------------------------
// Weight transpose: wT[(k9*COUT + n)*CIN + c] bf16 (WMMA B K-pairs contiguous)
// ---------------------------------------------------------------------------
template <int CIN, int COUT>
__global__ void prep_w_kernel(const float* __restrict__ w, bf16* __restrict__ wT) {
    int i = blockIdx.x * 256 + threadIdx.x;
    if (i >= 9 * CIN * COUT) return;
    int c  = i % CIN;
    int nk = i / CIN;
    int n  = nk % COUT;
    int k  = nk / COUT;
    wT[i] = (bf16)w[((size_t)(k * CIN + c)) * COUT + n];
}

__global__ void prep_as_kernel(const float* __restrict__ A, float* __restrict__ As, int total) {
    int i = blockIdx.x * 256 + threadIdx.x;
    if (i < total) As[i] = 0.9f * tanhf(A[i]);
}

// ---------------------------------------------------------------------------
// WMMA implicit-GEMM conv: k=3, s=2, SAME (pad_before=0, pad_after=1).
// 8 waves/block, each owns a 16(M-pixels)x16(N-ch) tile; K = 3*3*CIN in
// 32-wide steps of v_wmma_f32_16x16x32_bf16. Slab staged via async-LDS when
// the toolchain exposes it (interior tiles are 3 contiguous rows).
// ---------------------------------------------------------------------------
template <int CIN, int COUT>
__global__ __launch_bounds__(256)
void conv_wmma_kernel(const bf16* __restrict__ in, const bf16* __restrict__ wT,
                      const float* __restrict__ bias, bf16* __restrict__ out,
                      int Hin, int Win, int Hout, int Wout) {
    const int xtiles = Wout >> 4;
    const int xt = blockIdx.x % xtiles;
    const int nb = blockIdx.x / xtiles;
    const int y  = blockIdx.y;
    const int t  = blockIdx.z;
    const int x0 = xt << 4;

    const int tid  = threadIdx.x;
    const int wave = tid >> 5;
    const int lane = tid & 31;

    __shared__ __align__(16) bf16 slab[3 * 34 * CIN];

    // Pull the first weight tiles toward GL2 while we stage the input slab.
    __builtin_prefetch(&wT[(size_t)(nb * 128 + wave * 16) * CIN], 0, 1);

    const bool interior = (2 * y + 2 < Hin) && (2 * x0 + 33 < Win);
#if USE_ASYNC_LDS
    if (interior) {
        // 3 contiguous rows of 34*CIN bf16: bulk ASYNCcnt copies, 16B units.
        constexpr int perRow = (34 * CIN) / 8;           // b128 units per row
        for (int u = tid; u < 3 * perRow; u += 256) {
            int row = u / perRow;
            int k   = u - row * perRow;
            const bf16* g = &in[(((size_t)t * Hin + (2 * y + row)) * Win + 2 * x0) * CIN + k * 8];
            bf16* l = &slab[(row * 34) * CIN + k * 8];
            __builtin_amdgcn_global_load_async_to_lds_b128(
                (v4i*)(size_t)g, (v4i*)(size_t)l, 0, 0);
        }
        __builtin_amdgcn_s_wait_asynccnt(0);
        __syncthreads();
    } else
#endif
    {
        for (int i = tid; i < 3 * 34 * CIN; i += 256) {
            int c   = i % CIN;
            int col = (i / CIN) % 34;
            int row = i / (CIN * 34);
            int iy = 2 * y  + row;
            int ix = 2 * x0 + col;
            bf16 v = (bf16)0.0f;
            if (interior || (iy < Hin && ix < Win))
                v = in[(((size_t)t * Hin + iy) * Win + ix) * CIN + c];
            slab[i] = v;
        }
        __syncthreads();
    }

    const int hi  = lane >> 4;    // half-wave K sub-range (ISA 16x16x32 layout)
    const int l16 = lane & 15;
    const int n0  = nb * 128 + wave * 16;

    v8f acc = {};

    for (int ky = 0; ky < 3; ++ky)
        for (int kx = 0; kx < 3; ++kx) {
            const int colA = 2 * l16 + kx;
            const unsigned* slabU =
                (const unsigned*)&slab[((ky * 34) + colA) * CIN];
            const unsigned* wU =
                (const unsigned*)&wT[((size_t)((ky * 3 + kx) * COUT) + (n0 + l16)) * CIN];
            // Prefetch weights 1/16 of a (ky,kx) panel ahead of this lane's row.
            __builtin_prefetch(wU + CIN / 2 * COUT / 16, 0, 1);
#pragma unroll
            for (int cb = 0; cb < CIN / 32; ++cb) {
                v8u au, bu;
#pragma unroll
                for (int j = 0; j < 8; ++j) {
                    // A layout: vgpr j holds K-pair; kbase = 2j (+8 j>=4)(+8 hi)
                    int kbase = 2 * j + ((j >= 4) ? 8 : 0) + hi * 8;
                    int kidx  = (cb * 32 + kbase) >> 1;
                    au[j] = slabU[kidx];
                    bu[j] = wU[kidx];
                }
                v16bf a = __builtin_bit_cast(v16bf, au);
                v16bf b = __builtin_bit_cast(v16bf, bu);
                acc = __builtin_amdgcn_wmma_f32_16x16x32_bf16(
                    false, a, false, b, (short)0, acc, false, false);
            }
        }

    // D layout: lane -> N = l16; vgpr r -> M = r + hi*8 (output x = x0 + M)
    const int n = n0 + l16;
    const float bn_ = bias[n];
#pragma unroll
    for (int r = 0; r < 8; ++r) {
        int m = r + hi * 8;
        float v = acc[r] + bn_;
        v = v > 0.0f ? v : 0.0f;
        out[(((size_t)t * Hout + y) * Wout + (x0 + m)) * COUT + n] = (bf16)v;
    }
}

// ---------------------------------------------------------------------------
// Query features: nearest t, bilinear (y,x) on X3. grid 4, block 256.
// ---------------------------------------------------------------------------
__global__ void qfeat_kernel(const float* __restrict__ qp, const bf16* __restrict__ x3,
                             float* __restrict__ qf) {
    int n = blockIdx.x, c = threadIdx.x;
    float qt = qp[n * 3 + 0], qy = qp[n * 3 + 1], qx = qp[n * 3 + 2];
    int t = (int)(qt * 15.0f);
    t = t < 0 ? 0 : (t > 15 ? 15 : t);
    float yf = qy * 31.0f, xf = qx * 31.0f;
    int y0 = (int)floorf(yf); y0 = y0 < 0 ? 0 : (y0 > 31 ? 31 : y0);
    int y1 = y0 + 1 > 31 ? 31 : y0 + 1;
    int x0 = (int)floorf(xf); x0 = x0 < 0 ? 0 : (x0 > 31 ? 31 : x0);
    int x1 = x0 + 1 > 31 ? 31 : x0 + 1;
    float wy1 = yf - (float)y0, wy0 = 1.0f - wy1;
    float wx1 = xf - (float)x0, wx0 = 1.0f - wx1;
    const bf16* base = &x3[(size_t)t * 32 * 32 * 256 + c];
    float f00 = (float)base[((size_t)y0 * 32 + x0) * 256];
    float f01 = (float)base[((size_t)y0 * 32 + x1) * 256];
    float f10 = (float)base[((size_t)y1 * 32 + x0) * 256];
    float f11 = (float)base[((size_t)y1 * 32 + x1) * 256];
    qf[n * 256 + c] = (f00 * wx0 + f01 * wx1) * wy0 + (f10 * wx0 + f11 * wx1) * wy1;
}

// ---------------------------------------------------------------------------
// corr + softmax(h,w) + proj: CS[n,t,p,64]
// ---------------------------------------------------------------------------
__global__ __launch_bounds__(256)
void corr_proj_kernel(const float* __restrict__ qf, const bf16* __restrict__ x3,
                      const float* __restrict__ projw, const float* __restrict__ projb,
                      float* __restrict__ cs) {
    const int t = blockIdx.x, n = blockIdx.y, tid = threadIdx.x;
    __shared__ float logit[1024];
    __shared__ float red[256];
    const float* q = &qf[n * 256];

    for (int j = 0; j < 4; ++j) {
        int p = tid + 256 * j;
        const bf16* xp = &x3[((size_t)t * 1024 + p) * 256];
        float s = 0.0f;
        for (int c = 0; c < 256; ++c) s += (float)xp[c] * q[c];
        logit[p] = s * (1.0f / 16.0f);   // / sqrt(HIDDEN)
    }
    __syncthreads();

    float lm = -1e30f;
    for (int j = 0; j < 4; ++j) lm = fmaxf(lm, logit[tid + 256 * j]);
    float mx = block_reduce_max(lm, red, tid);

    float ls = 0.0f;
    for (int j = 0; j < 4; ++j) {
        float e = expf(logit[tid + 256 * j] - mx);
        logit[tid + 256 * j] = e;
        ls += e;
    }
    float tot = block_reduce_sum(ls, red, tid);
    float inv = 1.0f / tot;

    for (int j = 0; j < 4; ++j) {
        int p = tid + 256 * j;
        float a = logit[p] * inv;
        float* o = &cs[(((size_t)n * 16 + t) * 1024 + p) * 64];
        for (int c = 0; c < 64; ++c) o[c] = a * projw[c] + projb[c];
    }
}

// ---------------------------------------------------------------------------
// Depthwise CIRCULAR 3x7x7 conv (SSM geometric-series step):
//   out[p] = sum_taps K[c,tap] * in[(p - offset) mod (16,32,32)]  (+ bx)
// Slab = 3 t-planes x 32x32 x 4ch fp32 (49 KB). Staged by the Tensor Data
// Mover when available: 3 D# descriptors (tile 4 x 1024, row pitch 64 fp32),
// issued by wave 0, completion via s_wait_tensorcnt.
// ---------------------------------------------------------------------------
__global__ __launch_bounds__(256)
void dwconv_kernel(const float* __restrict__ in, const float* __restrict__ ker,
                   const float* __restrict__ bx, float* __restrict__ out, int addbx) {
    __shared__ float slab[3 * 1024 * 4];
    __shared__ float kl[4 * 147];
    const int cb = blockIdx.x, t = blockIdx.y, bn = blockIdx.z;
    const int c0 = cb * 4;
    const int tid = threadIdx.x;

    for (int i = tid; i < 4 * 147; i += 256)
        kl[i] = ker[(size_t)(c0 + i / 147) * 147 + (i % 147)];

#if USE_TDM
    if (tid < 32) {
        // One wave issues 3 tensor-DMA descriptors (one per wrapped t-plane).
#pragma unroll
        for (int s = 0; s < 3; ++s) {
            const int ts = (t + s + 15) & 15;
            unsigned long long ga = (unsigned long long)(size_t)
                &in[(((size_t)bn * 16 + ts) * 1024) * 64 + c0];
            unsigned lds = (unsigned)(size_t)&slab[s * 4096];
            // D# group0: count=1 | lds_addr | global_addr(57) | type=2
            v4u g0 = { 1u, lds,
                       (unsigned)(ga & 0xffffffffull),
                       (unsigned)((ga >> 32) & 0x1ffffffull) | (2u << 30) };
            // D# group1: data_size=4B(2)<<16 | tensor_dim0=64 @48 |
            //            tensor_dim1=1024 @80 | tile_dim0=4 @112 |
            //            tile_dim1=1024 @128 | tensor_dim0_stride=64 @160
            v8i g1 = { 0x00020000, 0x00400000, 0x04000000, 0x00040000,
                       0x00000400, 64, 0, 0 };
            v4i gz = { 0, 0, 0, 0 };
#if TDM_SIX_ARG
            v8i z8 = { 0, 0, 0, 0, 0, 0, 0, 0 };
            __builtin_amdgcn_tensor_load_to_lds(g0, g1, gz, gz, z8, 0);
#else
            __builtin_amdgcn_tensor_load_to_lds(g0, g1, gz, gz, 0);
#endif
        }
        __builtin_amdgcn_s_wait_tensorcnt(0);
    }
    __syncthreads();
#else
    for (int i = tid; i < 3 * 1024 * 4; i += 256) {
        int c = i & 3;
        int p = (i >> 2) & 1023;
        int s = i >> 12;                       // plane s holds t + s - 1 (mod 16)
        int ts = (t + s + 15) & 15;
        slab[i] = in[(((size_t)bn * 16 + ts) * 1024 + p) * 64 + c0 + c];
    }
    __syncthreads();
#endif

    int h[4], w[4];
#pragma unroll
    for (int j = 0; j < 4; ++j) {
        int p = tid + 256 * j;
        h[j] = p >> 5;
        w[j] = p & 31;
    }

    float acc[4][4] = {};
    for (int it = 0; it < 3; ++it) {
        const int s = 2 - it;                  // source t - (it-1)
        for (int ih = 0; ih < 7; ++ih)
            for (int iw = 0; iw < 7; ++iw) {
                const int tap = (it * 7 + ih) * 7 + iw;
                const float k0 = kl[0 * 147 + tap];
                const float k1 = kl[1 * 147 + tap];
                const float k2 = kl[2 * 147 + tap];
                const float k3 = kl[3 * 147 + tap];
#pragma unroll
                for (int j = 0; j < 4; ++j) {
                    int hh = (h[j] + 35 - ih) & 31;     // (h - (ih-3)) mod 32
                    int ww = (w[j] + 35 - iw) & 31;
                    const float* sp = &slab[((s << 10) + (hh << 5) + ww) * 4];
                    acc[j][0] += k0 * sp[0];
                    acc[j][1] += k1 * sp[1];
                    acc[j][2] += k2 * sp[2];
                    acc[j][3] += k3 * sp[3];
                }
            }
    }

#pragma unroll
    for (int j = 0; j < 4; ++j) {
        int p = tid + 256 * j;
        size_t base = (((size_t)bn * 16 + t) * 1024 + p) * 64 + c0;
        if (addbx) {
            const float* bp = &bx[base];
            for (int c = 0; c < 4; ++c) out[base + c] = acc[j][c] + bp[c];
        } else {
            for (int c = 0; c < 4; ++c) out[base + c] = acc[j][c];
        }
    }
}

// ---------------------------------------------------------------------------
// LayerNorm(C=64) of res, scale/bias, residual add into CS.
// ---------------------------------------------------------------------------
__global__ __launch_bounds__(256)
void ln_residual_kernel(const float* __restrict__ res, const float* __restrict__ scale,
                        const float* __restrict__ biasv, float* __restrict__ cs) {
    size_t pos = (size_t)blockIdx.x * 256 + threadIdx.x;   // 65536 positions
    const float* r = &res[pos * 64];
    float m = 0.0f;
    for (int c = 0; c < 64; ++c) m += r[c];
    m *= (1.0f / 64.0f);
    float v = 0.0f;
    for (int c = 0; c < 64; ++c) { float d = r[c] - m; v += d * d; }
    v *= (1.0f / 64.0f);
    float inv = rsqrtf(v + 1e-6f);
    float* o = &cs[pos * 64];
    for (int c = 0; c < 64; ++c)
        o[c] += (r[c] - m) * inv * scale[c] + biasv[c];
}

// ---------------------------------------------------------------------------
// Heads: heat -> softmax(x10) -> expected coords (*8); pooled -> occlusion.
// ---------------------------------------------------------------------------
__global__ __launch_bounds__(256)
void head_kernel(const float* __restrict__ cs, const float* __restrict__ heatw,
                 const float* __restrict__ heatb, const float* __restrict__ occw,
                 const float* __restrict__ occb, float* __restrict__ outp) {
    const int t = blockIdx.x, n = blockIdx.y, tid = threadIdx.x;
    __shared__ float heat[1024];
    __shared__ float red[256];
    __shared__ float chp[4][64];
    __shared__ float sh_se, sh_sx;

    const float hb = heatb[0];
    for (int j = 0; j < 4; ++j) {
        int p = tid + 256 * j;
        const float* f = &cs[(((size_t)n * 16 + t) * 1024 + p) * 64];
        float s = hb;
        for (int c = 0; c < 64; ++c) s += f[c] * heatw[c];
        heat[p] = s;
    }
    __syncthreads();

    float lm = -1e30f;
    for (int j = 0; j < 4; ++j) lm = fmaxf(lm, heat[tid + 256 * j]);
    float mx = block_reduce_max(lm, red, tid);

    float se = 0.0f, sx = 0.0f, sy = 0.0f;
    for (int j = 0; j < 4; ++j) {
        int p = tid + 256 * j;
        float e = expf(10.0f * (heat[p] - mx));
        se += e;
        sx += e * (float)(p & 31);   // x coord = w
        sy += e * (float)(p >> 5);   // y coord = h
    }
    float tse = block_reduce_sum(se, red, tid);
    if (tid == 0) sh_se = tse;
    float tsx = block_reduce_sum(sx, red, tid);
    if (tid == 0) sh_sx = tsx;
    float tsy = block_reduce_sum(sy, red, tid);
    if (tid == 0) {
        float invs = 1.0f / sh_se;
        outp[((n * 16 + t) * 2) + 0] = sh_sx * invs * 8.0f;   // * (W/fW)
        outp[((n * 16 + t) * 2) + 1] = tsy   * invs * 8.0f;   // * (H/fH)
    }
    __syncthreads();

    const int c = tid & 63, g = tid >> 6;
    float s = 0.0f;
    for (int p = g; p < 1024; p += 4)
        s += cs[(((size_t)n * 16 + t) * 1024 + p) * 64 + c];
    chp[g][c] = s;
    __syncthreads();
    if (tid < 64)
        red[tid] = (chp[0][tid] + chp[1][tid] + chp[2][tid] + chp[3][tid]) *
                   (1.0f / 1024.0f) * occw[tid];
    __syncthreads();
    if (tid == 0) {
        float acc = occb[0];
        for (int cc = 0; cc < 64; ++cc) acc += red[cc];
        outp[128 + n * 16 + t] = 1.0f / (1.0f + expf(-acc));
    }
}

// ---------------------------------------------------------------------------
// Host side
// ---------------------------------------------------------------------------
extern "C" void kernel_launch(void* const* d_in, const int* in_sizes, int n_in,
                              void* d_out, int out_size, void* d_ws, size_t ws_size,
                              hipStream_t stream) {
    (void)in_sizes; (void)n_in; (void)out_size; (void)ws_size;

    const float* video = (const float*)d_in[0];
    const float* qp    = (const float*)d_in[1];
    const float* w1    = (const float*)d_in[2];
    const float* b1    = (const float*)d_in[3];
    const float* w2    = (const float*)d_in[4];
    const float* b2    = (const float*)d_in[5];
    const float* w3    = (const float*)d_in[6];
    const float* b3    = (const float*)d_in[7];
    const float* projw = (const float*)d_in[8];
    const float* projb = (const float*)d_in[9];
    const float* Ak    = (const float*)d_in[10];
    const float* Bk    = (const float*)d_in[11];
    const float* lns   = (const float*)d_in[12];
    const float* lnb   = (const float*)d_in[13];
    const float* heatw = (const float*)d_in[14];
    const float* heatb = (const float*)d_in[15];
    const float* occw  = (const float*)d_in[16];
    const float* occb  = (const float*)d_in[17];
    float* out = (float*)d_out;

    size_t off = 0;
    auto carve = [&](size_t bytes) -> char* {
        off = (off + 255) & ~(size_t)255;
        char* p = (char*)d_ws + off;
        off += bytes;
        return p;
    };
    bf16*  X1  = (bf16*)carve((size_t)16 * 128 * 128 * 64 * 2);
    bf16*  X2  = (bf16*)carve((size_t)16 * 64 * 64 * 128 * 2);
    bf16*  X3  = (bf16*)carve((size_t)16 * 32 * 32 * 256 * 2);
    bf16*  W2T = (bf16*)carve((size_t)9 * 64 * 128 * 2);
    bf16*  W3T = (bf16*)carve((size_t)9 * 128 * 256 * 2);
    float* AS  = (float*)carve((size_t)3 * 64 * 147 * 4);
    float* QF  = (float*)carve((size_t)4 * 256 * 4);
    float* CS  = (float*)carve((size_t)4 * 16 * 1024 * 64 * 4);
    float* BX  = (float*)carve((size_t)4 * 16 * 1024 * 64 * 4);
    float* S0  = (float*)carve((size_t)4 * 16 * 1024 * 64 * 4);
    float* S1  = (float*)carve((size_t)4 * 16 * 1024 * 64 * 4);

    conv1_kernel<<<dim3(8, 128, 16), 256, 0, stream>>>(video, w1, b1, X1);
    prep_w_kernel<64, 128><<<(9 * 64 * 128 + 255) / 256, 256, 0, stream>>>(w2, W2T);
    prep_w_kernel<128, 256><<<(9 * 128 * 256 + 255) / 256, 256, 0, stream>>>(w3, W3T);
    prep_as_kernel<<<(3 * 64 * 147 + 255) / 256, 256, 0, stream>>>(Ak, AS, 3 * 64 * 147);

    conv_wmma_kernel<64, 128><<<dim3(4 * 1, 64, 16), 256, 0, stream>>>(
        X1, W2T, b2, X2, 128, 128, 64, 64);
    conv_wmma_kernel<128, 256><<<dim3(2 * 2, 32, 16), 256, 0, stream>>>(
        X2, W3T, b3, X3, 64, 64, 32, 32);

    qfeat_kernel<<<4, 256, 0, stream>>>(qp, X3, QF);
    corr_proj_kernel<<<dim3(16, 4), 256, 0, stream>>>(QF, X3, projw, projb, CS);

    for (int r = 0; r < 3; ++r) {
        const float* Bker = Bk + (size_t)r * 64 * 147;
        const float* Aker = AS + (size_t)r * 64 * 147;
        dwconv_kernel<<<dim3(16, 16, 4), 256, 0, stream>>>(CS, Bker, CS, BX, 0);
        for (int k = 0; k < 7; ++k) {
            const float* src = (k == 0) ? BX : ((k & 1) ? S0 : S1);
            float* dst = (k & 1) ? S1 : S0;
            dwconv_kernel<<<dim3(16, 16, 4), 256, 0, stream>>>(src, Aker, BX, dst, 1);
        }
        ln_residual_kernel<<<256, 256, 0, stream>>>(S0, lns + r * 64, lnb + r * 64, CS);
    }

    head_kernel<<<dim3(16, 4), 256, 0, stream>>>(CS, heatw, heatb, occw, occb, out);
}
